// CrossTransformerLayer_88252987998700
// MI455X (gfx1250) — matrix operand order repaired and checked by reading, not compiled, not run on hardware
//
#include <hip/hip_runtime.h>

// ---------------------------------------------------------------------------
// CDNA5 (gfx1250) implementation of CrossTransformerLayer.
// All GEMMs use v_wmma_f32_16x16x32_f16 (f16 in, f32 accumulate).
// Weights are pre-transposed to f16 Bt[N][K] so both A and B WMMA fragments
// are contiguous 16-byte loads per lane (no LDS staging in GEMM).
// ---------------------------------------------------------------------------

typedef __attribute__((ext_vector_type(16))) _Float16 v16h;
typedef __attribute__((ext_vector_type(8)))  _Float16 v8h;
typedef __attribute__((ext_vector_type(8)))  float    v8f;

union H16 { v16h v; v8h h2[2]; _Float16 e[16]; };

__device__ inline v8f vzero8() {
  v8f z;
#pragma unroll
  for (int i = 0; i < 8; ++i) z[i] = 0.f;
  return z;
}

__device__ inline v8f wmma_f16(const H16& a, const H16& b, v8f c) {
  return __builtin_amdgcn_wmma_f32_16x16x32_f16(false, a.v, false, b.v,
                                                (short)0, c, false, false);
}

// --------------------------- elementwise helpers ---------------------------

__global__ void f32_to_f16(const float* __restrict__ in, _Float16* __restrict__ out, int n) {
  int i = blockIdx.x * 256 + threadIdx.x;
  if (i < n) out[i] = (_Float16)in[i];
}

// in: f32 [K][N] row-major  ->  out: f16 [N][K] row-major (Bt layout)
__global__ void transpose_f32_to_f16(const float* __restrict__ in, _Float16* __restrict__ out,
                                     int K, int N) {
  int i = blockIdx.x * 256 + threadIdx.x;
  if (i < K * N) {
    int n = i / K, k = i % K;
    out[i] = (_Float16)in[(size_t)k * N + n];
  }
}

// ------------------------------- WMMA GEMM ---------------------------------
// C[M,N] = A[M,K](f16) x Bt[N,K](f16)^T + bias[N]; optional relu;
// writes f32 (Cf) and/or f16 (Ch; trans16 -> Ch[N][M]).
// Block: 256 threads = 8 waves (4 M-tiles x 2 N-tiles of 16x32 per wave).
__global__ __launch_bounds__(256)
void gemm_wmma(const _Float16* __restrict__ A, const _Float16* __restrict__ Bt,
               const float* __restrict__ bias, float* __restrict__ Cf,
               _Float16* __restrict__ Ch, int M, int N, int K,
               int trans16, int relu) {
  const int lane = threadIdx.x & 31;
  const int wv   = threadIdx.x >> 5;
  const int wm   = wv & 3;        // 0..3 -> M tile
  const int wn   = wv >> 2;       // 0..1 -> N tile (32 wide)
  const int lm   = lane & 15;
  const int hi   = lane >> 4;
  const int gm0  = blockIdx.x * 64 + wm * 16;
  const int gn0  = blockIdx.y * 64 + wn * 32;

  v8f acc0 = vzero8(), acc1 = vzero8();

  const _Float16* ap  = A  + (size_t)(gm0 + lm) * K + hi * 16;       // A row, K-chunk hi*16
  const _Float16* bp0 = Bt + (size_t)(gn0 + lm) * K + hi * 16;       // B col n, K-chunk hi*16
  const _Float16* bp1 = Bt + (size_t)(gn0 + 16 + lm) * K + hi * 16;

  for (int k0 = 0; k0 < K; k0 += 32) {
    H16 fa, fb0, fb1;
    fa.h2[0]  = *(const v8h*)(ap  + k0);
    fa.h2[1]  = *(const v8h*)(ap  + k0 + 8);
    fb0.h2[0] = *(const v8h*)(bp0 + k0);
    fb0.h2[1] = *(const v8h*)(bp0 + k0 + 8);
    fb1.h2[0] = *(const v8h*)(bp1 + k0);
    fb1.h2[1] = *(const v8h*)(bp1 + k0 + 8);
    acc0 = wmma_f16(fa, fb0, acc0);
    acc1 = wmma_f16(fa, fb1, acc1);
  }

#pragma unroll
  for (int nt = 0; nt < 2; ++nt) {
    v8f a = nt ? acc1 : acc0;
    int gn = gn0 + nt * 16 + lm;
    float bb = bias[gn];
#pragma unroll
    for (int v = 0; v < 8; ++v) {
      int gm = gm0 + v + 8 * hi;
      float val = a[v] + bb;
      if (relu) val = fmaxf(val, 0.f);
      if (Cf) Cf[(size_t)gm * N + gn] = val;
      if (Ch) {
        if (trans16) Ch[(size_t)gn * M + gm] = (_Float16)val;
        else         Ch[(size_t)gm * N + gn] = (_Float16)val;
      }
    }
  }
}

// --------------------------- cross attention -------------------------------
// grid: 2048 blocks = (pair p in 0..255) x (head h in 0..7); 64 threads = 2 waves.
// Each wave handles query tiles {w, w+2, ..., w+10}. Flash-style: scores for a
// full 16x192 row kept in 12 WMMA accumulators, softmax via lane shuffles,
// probabilities staged in LDS to convert C-layout -> A-layout for P x V.
__global__ __launch_bounds__(64)
void cross_attn(const _Float16* __restrict__ Qc, const _Float16* __restrict__ Kc,
                const _Float16* __restrict__ Vt, const int* __restrict__ mask,
                _Float16* __restrict__ O1) {
  __shared__ _Float16 P[2][16][200];   // per-wave prob tile, 16B-aligned rows

  const int p = blockIdx.x >> 3;
  const int h = blockIdx.x & 7;
  const int i = p >> 4;                // query sequence
  const int j = p & 15;                // key/value sequence
  const int w    = threadIdx.x >> 5;
  const int lane = threadIdx.x & 31;
  const int lm   = lane & 15;
  const int hi   = lane >> 4;

  // per-column mask additive term (cols n = lm, key tile kt)
  float madd[12];
#pragma unroll
  for (int kt = 0; kt < 12; ++kt)
    madd[kt] = (mask[p * 192 + kt * 16 + lm] != 0) ? 0.f : -1e9f;

  for (int qt = w; qt < 12; qt += 2) {
    // --- Q fragments (A-matrix), K = 64 -> 2 fragments
    const _Float16* qbase = Qc + (size_t)(i * 192 + qt * 16 + lm) * 512 + h * 64;
    H16 aq0, aq1;
    aq0.h2[0] = *(const v8h*)(qbase + hi * 16);
    aq0.h2[1] = *(const v8h*)(qbase + hi * 16 + 8);
    aq1.h2[0] = *(const v8h*)(qbase + 32 + hi * 16);
    aq1.h2[1] = *(const v8h*)(qbase + 32 + hi * 16 + 8);

    // --- scores: 12 key tiles x (2 WMMA over K=64). B = K^T, so each lane
    //     reads 16 contiguous head-dims of one key row.
    v8f sacc[12];
#pragma unroll
    for (int kt = 0; kt < 12; ++kt) {
      sacc[kt] = vzero8();
      const _Float16* kbase = Kc + (size_t)(j * 192 + kt * 16 + lm) * 512 + h * 64;
      H16 bk0, bk1;
      bk0.h2[0] = *(const v8h*)(kbase + hi * 16);
      bk0.h2[1] = *(const v8h*)(kbase + hi * 16 + 8);
      bk1.h2[0] = *(const v8h*)(kbase + 32 + hi * 16);
      bk1.h2[1] = *(const v8h*)(kbase + 32 + hi * 16 + 8);
      sacc[kt] = wmma_f16(aq0, bk0, sacc[kt]);
      sacc[kt] = wmma_f16(aq1, bk1, sacc[kt]);
    }

    // --- softmax per row m = v + 8*hi (row spread over the 16-lane half)
#pragma unroll
    for (int v = 0; v < 8; ++v) {
      float mx = -3.4e38f;
#pragma unroll
      for (int kt = 0; kt < 12; ++kt) {
        float t = sacc[kt][v] * 0.125f + madd[kt];   // /sqrt(64) + mask
        sacc[kt][v] = t;
        mx = fmaxf(mx, t);
      }
      for (int d = 1; d < 16; d <<= 1) mx = fmaxf(mx, __shfl_xor(mx, d, 32));
      float sum = 0.f;
#pragma unroll
      for (int kt = 0; kt < 12; ++kt) {
        float e = __expf(sacc[kt][v] - mx);
        sacc[kt][v] = e;
        sum += e;
      }
      for (int d = 1; d < 16; d <<= 1) sum += __shfl_xor(sum, d, 32);
      float inv = 1.f / sum;
      int m = v + 8 * hi;
#pragma unroll
      for (int kt = 0; kt < 12; ++kt)
        P[w][m][kt * 16 + lm] = (_Float16)(sacc[kt][v] * inv);
    }

    // --- O = P[16x192] x V[192x64]; B read from transposed Vt[512][3072]
    v8f oacc[4];
#pragma unroll
    for (int nt = 0; nt < 4; ++nt) oacc[nt] = vzero8();
#pragma unroll
    for (int ks = 0; ks < 6; ++ks) {
      H16 apf;
      apf.h2[0] = *(const v8h*)&P[w][lm][ks * 32 + hi * 16];
      apf.h2[1] = *(const v8h*)&P[w][lm][ks * 32 + hi * 16 + 8];
#pragma unroll
      for (int nt = 0; nt < 4; ++nt) {
        int gn = h * 64 + nt * 16 + lm;
        const _Float16* vp = Vt + (size_t)gn * 3072 + j * 192 + ks * 32 + hi * 16;
        H16 bv;
        bv.h2[0] = *(const v8h*)(vp);
        bv.h2[1] = *(const v8h*)(vp + 8);
        oacc[nt] = wmma_f16(apf, bv, oacc[nt]);
      }
    }

    // --- store context (f16) into O1[p*192 + l][h*64 + d]
#pragma unroll
    for (int nt = 0; nt < 4; ++nt) {
      int gn = h * 64 + nt * 16 + lm;
#pragma unroll
      for (int v = 0; v < 8; ++v) {
        int gm = p * 192 + qt * 16 + v + 8 * hi;
        O1[(size_t)gm * 512 + gn] = (_Float16)oacc[nt][v];
      }
    }
  }
}

// ---------------------------- inner attention ------------------------------
// 3072 batches (r = j*192 + l), Lq=1, Lk=16 keys (key t -> row (t*16+j)*192+l).
// Tiny (50 MFLOP): plain VALU. Block = 64 threads (one per head-dim).
__global__ __launch_bounds__(64)
void inner_attn(const _Float16* __restrict__ Qin, const _Float16* __restrict__ Kin,
                const _Float16* __restrict__ Vin, _Float16* __restrict__ O2) {
  __shared__ float qs[64];
  __shared__ float as[16];
  const int r = blockIdx.x;
  const int j = r / 192, l = r % 192;
  const int d = threadIdx.x;

  for (int h = 0; h < 8; ++h) {
    qs[d] = (float)Qin[(size_t)r * 512 + h * 64 + d];
    __syncthreads();
    if (d < 16) {
      int kr = (d * 16 + j) * 192 + l;
      const _Float16* kp = Kin + (size_t)kr * 512 + h * 64;
      float s = 0.f;
      for (int dd = 0; dd < 64; ++dd) s += qs[dd] * (float)kp[dd];
      as[d] = s * 0.125f;
    }
    __syncthreads();
    float mx = -3.4e38f;
    for (int t = 0; t < 16; ++t) mx = fmaxf(mx, as[t]);
    float wgt[16], sum = 0.f;
    for (int t = 0; t < 16; ++t) { wgt[t] = __expf(as[t] - mx); sum += wgt[t]; }
    float o = 0.f;
    for (int t = 0; t < 16; ++t) {
      int kr = (t * 16 + j) * 192 + l;
      o += wgt[t] * (float)Vin[(size_t)kr * 512 + h * 64 + d];
    }
    O2[(size_t)r * 512 + h * 64 + d] = (_Float16)(o / sum);
    __syncthreads();
  }
}

// -------------------------- residual + layernorm ---------------------------
// x = A + B, y = (x - mu)/sqrt(var + 1e-6)*g + be; writes f32 and optional f16.
__global__ __launch_bounds__(128)
void add_layernorm(const float* __restrict__ A, const float* __restrict__ B,
                   const float* __restrict__ g, const float* __restrict__ be,
                   float* __restrict__ outF, _Float16* __restrict__ outH) {
  __shared__ float red[128];
  const int r = blockIdx.x, t = threadIdx.x;
  float x[4];
#pragma unroll
  for (int u = 0; u < 4; ++u) {
    int c = t + u * 128;
    x[u] = A[(size_t)r * 512 + c] + B[(size_t)r * 512 + c];
  }
  float s = x[0] + x[1] + x[2] + x[3];
  red[t] = s; __syncthreads();
  for (int o = 64; o > 0; o >>= 1) { if (t < o) red[t] += red[t + o]; __syncthreads(); }
  float mu = red[0] / 512.f;
  __syncthreads();
  float vs = 0.f;
#pragma unroll
  for (int u = 0; u < 4; ++u) { float dd = x[u] - mu; vs += dd * dd; }
  red[t] = vs; __syncthreads();
  for (int o = 64; o > 0; o >>= 1) { if (t < o) red[t] += red[t + o]; __syncthreads(); }
  float inv = rsqrtf(red[0] / 512.f + 1e-6f);
#pragma unroll
  for (int u = 0; u < 4; ++u) {
    int c = t + u * 128;
    float y = (x[u] - mu) * inv * g[c] + be[c];
    if (outF) outF[(size_t)r * 512 + c] = y;
    if (outH) outH[(size_t)r * 512 + c] = (_Float16)y;
  }
}

// ------------------------------- launcher ----------------------------------

extern "C" void kernel_launch(void* const* d_in, const int* in_sizes, int n_in,
                              void* d_out, int out_size, void* d_ws, size_t ws_size,
                              hipStream_t stream) {
  (void)in_sizes; (void)n_in; (void)out_size; (void)ws_size;
  const float* S     = (const float*)d_in[0];
  const int*   mask  = (const int*)d_in[1];
  const float* ca_wq = (const float*)d_in[3];  const float* ca_bq = (const float*)d_in[4];
  const float* ca_wk = (const float*)d_in[5];  const float* ca_bk = (const float*)d_in[6];
  const float* ca_wv = (const float*)d_in[7];  const float* ca_bv = (const float*)d_in[8];
  const float* ca_wo = (const float*)d_in[9];  const float* ca_bo = (const float*)d_in[10];
  const float* ia_wq = (const float*)d_in[11]; const float* ia_bq = (const float*)d_in[12];
  const float* ia_wk = (const float*)d_in[13]; const float* ia_bk = (const float*)d_in[14];
  const float* ia_wv = (const float*)d_in[15]; const float* ia_bv = (const float*)d_in[16];
  const float* ia_wo = (const float*)d_in[17]; const float* ia_bo = (const float*)d_in[18];
  const float* fw1   = (const float*)d_in[19]; const float* fb1   = (const float*)d_in[20];
  const float* fw2   = (const float*)d_in[21]; const float* fb2   = (const float*)d_in[22];
  const float* ln1g  = (const float*)d_in[23]; const float* ln1b  = (const float*)d_in[24];
  const float* ln2g  = (const float*)d_in[25]; const float* ln2b  = (const float*)d_in[26];

  const size_t R = 3072, RP = 49152, Dm = 512, DF = 2048;

  char* wsp = (char*)d_ws;
  auto alloc = [&](size_t bytes) {
    char* pt = wsp; wsp += (bytes + 255) & ~(size_t)255; return pt;
  };
  _Float16* S16   = (_Float16*)alloc(R  * Dm * 2);
  _Float16* wq_t  = (_Float16*)alloc(Dm * Dm * 2);
  _Float16* wk_t  = (_Float16*)alloc(Dm * Dm * 2);
  _Float16* wv_t  = (_Float16*)alloc(Dm * Dm * 2);
  _Float16* wo_t  = (_Float16*)alloc(Dm * Dm * 2);
  _Float16* iq_t  = (_Float16*)alloc(Dm * Dm * 2);
  _Float16* ik_t  = (_Float16*)alloc(Dm * Dm * 2);
  _Float16* iv_t  = (_Float16*)alloc(Dm * Dm * 2);
  _Float16* io_t  = (_Float16*)alloc(Dm * Dm * 2);
  _Float16* w1_t  = (_Float16*)alloc(DF * Dm * 2);
  _Float16* w2_t  = (_Float16*)alloc(Dm * DF * 2);
  _Float16* Qc    = (_Float16*)alloc(R  * Dm * 2);
  _Float16* Kc    = (_Float16*)alloc(R  * Dm * 2);
  _Float16* Vt    = (_Float16*)alloc(Dm * R  * 2);   // transposed V: [512][3072]
  _Float16* O1    = (_Float16*)alloc(RP * Dm * 2);   // reused later as Kin
  _Float16* SA1   = (_Float16*)alloc(RP * Dm * 2);
  _Float16* Qin   = (_Float16*)alloc(R  * Dm * 2);
  _Float16* Vin   = (_Float16*)alloc(RP * Dm * 2);
  _Float16* O2    = (_Float16*)alloc(R  * Dm * 2);
  float*    SA2   = (float*)   alloc(R  * Dm * 4);
  float*    Snorm = (float*)   alloc(R  * Dm * 4);
  _Float16* Sn16  = (_Float16*)alloc(R  * Dm * 2);
  _Float16* H1    = (_Float16*)alloc(R  * DF * 2);
  float*    F2    = (float*)   alloc(R  * Dm * 4);
  _Float16* Kin   = O1;  // O1 is dead after the out-projection GEMM

  // 1) precision conversion + weight transposes
  {
    int n = (int)(R * Dm);
    f32_to_f16<<<(n + 255) / 256, 256, 0, stream>>>(S, S16, n);
    int nd = (int)(Dm * Dm), gd = (nd + 255) / 256;
    transpose_f32_to_f16<<<gd, 256, 0, stream>>>(ca_wq, wq_t, 512, 512);
    transpose_f32_to_f16<<<gd, 256, 0, stream>>>(ca_wk, wk_t, 512, 512);
    transpose_f32_to_f16<<<gd, 256, 0, stream>>>(ca_wv, wv_t, 512, 512);
    transpose_f32_to_f16<<<gd, 256, 0, stream>>>(ca_wo, wo_t, 512, 512);
    transpose_f32_to_f16<<<gd, 256, 0, stream>>>(ia_wq, iq_t, 512, 512);
    transpose_f32_to_f16<<<gd, 256, 0, stream>>>(ia_wk, ik_t, 512, 512);
    transpose_f32_to_f16<<<gd, 256, 0, stream>>>(ia_wv, iv_t, 512, 512);
    transpose_f32_to_f16<<<gd, 256, 0, stream>>>(ia_wo, io_t, 512, 512);
    int nf = (int)(Dm * DF), gf = (nf + 255) / 256;
    transpose_f32_to_f16<<<gf, 256, 0, stream>>>(fw1, w1_t, 512, 2048);
    transpose_f32_to_f16<<<gf, 256, 0, stream>>>(fw2, w2_t, 2048, 512);
  }

  dim3 gR (R  / 64, Dm / 64);   // [3072 x 512]
  dim3 gRP(RP / 64, Dm / 64);   // [49152 x 512]
  dim3 gF1(R  / 64, DF / 64);   // [3072 x 2048]

  // 2) shared Q/K/V projections (Q,K per sequence; V written transposed)
  gemm_wmma<<<gR, 256, 0, stream>>>(S16, wq_t, ca_bq, nullptr, Qc, (int)R, 512, 512, 0, 0);
  gemm_wmma<<<gR, 256, 0, stream>>>(S16, wk_t, ca_bk, nullptr, Kc, (int)R, 512, 512, 0, 0);
  gemm_wmma<<<gR, 256, 0, stream>>>(S16, wv_t, ca_bv, nullptr, Vt, (int)R, 512, 512, 1, 0);

  // 3) 256-pair x 8-head flash-style attention
  cross_attn<<<2048, 64, 0, stream>>>(Qc, Kc, Vt, mask, O1);

  // 4) output projection of cross attention
  gemm_wmma<<<gRP, 256, 0, stream>>>(O1, wo_t, ca_bo, nullptr, SA1, (int)RP, 512, 512, 0, 0);

  // 5) inner-attention projections (reshape is just a row permutation -> index in kernel)
  gemm_wmma<<<gR,  256, 0, stream>>>(S16, iq_t, ia_bq, nullptr, Qin, (int)R,  512, 512, 0, 0);
  gemm_wmma<<<gRP, 256, 0, stream>>>(SA1, ik_t, ia_bk, nullptr, Kin, (int)RP, 512, 512, 0, 0);
  gemm_wmma<<<gRP, 256, 0, stream>>>(SA1, iv_t, ia_bv, nullptr, Vin, (int)RP, 512, 512, 0, 0);

  // 6) inner attention (Lq=1, Lk=16) + output projection
  inner_attn<<<3072, 64, 0, stream>>>(Qin, Kin, Vin, O2);
  gemm_wmma<<<gR, 256, 0, stream>>>(O2, io_t, ia_bo, SA2, nullptr, (int)R, 512, 512, 0, 0);

  // 7) residual + LN1, FFN, residual + LN2
  add_layernorm<<<3072, 128, 0, stream>>>(S, SA2, ln1g, ln1b, Snorm, Sn16);
  gemm_wmma<<<gF1, 256, 0, stream>>>(Sn16, w1_t, fb1, nullptr, H1, (int)R, 2048, 512, 0, 1);
  gemm_wmma<<<gR,  256, 0, stream>>>(H1,  w2_t, fb2, F2, nullptr, (int)R, 512, 2048, 0, 0);
  add_layernorm<<<3072, 128, 0, stream>>>(Snorm, F2, ln2g, ln2b, (float*)d_out, nullptr);
}
